// tNet_59072980189448
// MI455X (gfx1250) — matrix operand order repaired and checked by reading, not compiled, not run on hardware
//
#include <hip/hip_runtime.h>
#include <hip/hip_bf16.h>

typedef __attribute__((ext_vector_type(16))) _Float16 v16h;
typedef __attribute__((ext_vector_type(8)))  _Float16 v8h;
typedef __attribute__((ext_vector_type(8)))  float    v8f;

// ---------------------------------------------------------------------------
// Operator composition (all f32, ~20 MFLOP total, fully parallel):
//   C[e][p][q]            = sum_f b2[e][p][f] * b1[f][q]
//   T1[a][d][p][q]        = sum_e u1[a][d][e] * C[e][p][q]
//   T2[a][d][b][g][q]     = sum_p T1[a][d][p][q] * u2[b][g][p]
//   W[d*128+g*16+r][a*128+b*16+c] = sum_q T2[a][d][b][g][q] * u3[c][r][q]
// Then: out = relu(x @ W^T + bias1) @ fc2_w^T + fc2_b
// ---------------------------------------------------------------------------

__global__ __launch_bounds__(256) void k_core(const float* __restrict__ b2,
                                              const float* __restrict__ b1,
                                              float* __restrict__ C) {
  int idx = blockIdx.x * 256 + threadIdx.x;            // 4096 elements
  if (idx >= 4096) return;
  int e = idx >> 8, p = (idx >> 4) & 15, q = idx & 15;
  const float* b2r = b2 + e * 256 + p * 16;
  float s = 0.f;
#pragma unroll
  for (int f = 0; f < 16; ++f) s += b2r[f] * b1[f * 16 + q];
  C[idx] = s;
}

__global__ __launch_bounds__(256) void k_t1(const float* __restrict__ u1,
                                            const float* __restrict__ C,
                                            float* __restrict__ T1) {
  int idx = blockIdx.x * 256 + threadIdx.x;            // 16384 elements
  if (idx >= 16384) return;
  int q = idx & 15, p = (idx >> 4) & 15, d = (idx >> 8) & 7, a = idx >> 11;
  const float* u1r = u1 + a * 128 + d * 16;            // over e
  float s = 0.f;
#pragma unroll
  for (int e = 0; e < 16; ++e) s += u1r[e] * C[e * 256 + p * 16 + q];
  T1[idx] = s;                                          // layout ((a*8+d)*16+p)*16+q
}

__global__ __launch_bounds__(256) void k_t2(const float* __restrict__ T1,
                                            const float* __restrict__ u2,
                                            float* __restrict__ T2) {
  int idx = blockIdx.x * 256 + threadIdx.x;            // 65536 elements
  if (idx >= 65536) return;
  int q = idx & 15, g = (idx >> 4) & 7, b = (idx >> 7) & 7;
  int d = (idx >> 10) & 7, a = idx >> 13;
  const float* t1r = T1 + ((a * 8 + d) * 16) * 16 + q; // stride 16 over p
  const float* u2r = u2 + b * 128 + g * 16;            // over p
  float s = 0.f;
#pragma unroll
  for (int p = 0; p < 16; ++p) s += t1r[p * 16] * u2r[p];
  T2[idx] = s;                                          // layout (((a*8+d)*8+b)*8+g)*16+q
}

__global__ __launch_bounds__(256) void k_w(const float* __restrict__ T2,
                                           const float* __restrict__ u3,
                                           _Float16* __restrict__ Wh) {
  int idx = blockIdx.x * 256 + threadIdx.x;            // 1M elements: idx = m*1024+n
  int m = idx >> 10, n = idx & 1023;
  int d = m >> 7, g = (m >> 4) & 7, r = m & 15;
  int a = n >> 7, b = (n >> 4) & 7, c = n & 15;
  const float* t2r = T2 + (((a * 8 + d) * 8 + b) * 8 + g) * 16;  // over q (contig)
  const float* u3r = u3 + c * 256 + r * 16;                      // over q (contig)
  float s = 0.f;
#pragma unroll
  for (int q = 0; q < 16; ++q) s += t2r[q] * u3r[q];
  Wh[idx] = (_Float16)s;                                // row-major [m][n]
}

__global__ __launch_bounds__(256) void k_cvtx(const float* __restrict__ x,
                                              _Float16* __restrict__ xh) {
  int idx = blockIdx.x * 256 + threadIdx.x;            // 1M elements
  xh[idx] = (_Float16)x[idx];
}

// Load a 16-half fragment as two b128 loads; `off2` = element offset of the
// second 8-half chunk (8 for A fragments, 8 for B fragments).
__device__ __forceinline__ v16h load_frag(const _Float16* __restrict__ p, int off2) {
  v8h lo = *(const v8h*)(p);
  v8h hi = *(const v8h*)(p + off2);
  return __builtin_shufflevector(lo, hi, 0, 1, 2, 3, 4, 5, 6, 7,
                                         8, 9, 10, 11, 12, 13, 14, 15);
}

// ---------------------------------------------------------------------------
// Main GEMM: h = relu(x @ W^T + bias1), fused fc2: out = h @ fc2_w^T + fc2_b.
// 64 blocks x 8 waves; block owns 16 batch rows, wave owns a 16x128 strip.
// A frag (16x32 f16): lane holds row z = zt + (lane&15), K chunk
//   [kb + hi*16 .. +15]                               (hi = lane>>4)
// B frag (32x16 f16): lane holds col m, elems i = W[m][kb + hi*16 + i]
// Fragments are software-pipelined one tile ahead so the B loads for tile t+1
// (and the A load for the next K block) are in flight during WMMA t — the
// wait before each WMMA relaxes from loadcnt==0 to loadcnt<=2.
// ---------------------------------------------------------------------------
__global__ __launch_bounds__(256) void k_gemm(const _Float16* __restrict__ xh,
                                              const _Float16* __restrict__ wh,
                                              const float* __restrict__ bias1,
                                              const float* __restrict__ fc2w,
                                              const float* __restrict__ fc2b,
                                              float* __restrict__ out) {
  __shared__ _Float16 lds_h[16 * 1024];                // 32 KB
  const int tid  = threadIdx.x;
  const int wave = tid >> 5;
  const int lane = tid & 31;
  const int col  = lane & 15;
  const int hi   = lane >> 4;
  const int zt   = blockIdx.x * 16;
  const int mbase = wave * 128;

  v8f acc[8] = {};
  const _Float16* arow  = xh + (zt + col) * 1024 + hi * 16;
  const _Float16* bbase = wh + (mbase + col) * 1024 + hi * 16; // tile t: +t*16*1024

  v16h a = load_frag(arow, 8);
  v16h b = load_frag(bbase, 8);

  for (int kb = 0; kb < 1024; kb += 32) {
    const int kn = (kb + 32) & 1023;                   // wrap: last prefetch is a
    v16h a_next = a;                                   // harmless redundant load
#pragma unroll
    for (int t = 0; t < 8; ++t) {
      v16h b_next;
      if (t == 7) {
        a_next = load_frag(arow + kn, 8);
        b_next = load_frag(bbase + kn, 8);
      } else {
        b_next = load_frag(bbase + (t + 1) * 16 * 1024 + kb, 8);
      }
      acc[t] = __builtin_amdgcn_wmma_f32_16x16x32_f16(
          /*neg_a=*/false, a, /*neg_b=*/false, b,
          /*c_mod=*/(short)0, acc[t], /*reuse_a=*/false, /*reuse_b=*/false);
      b = b_next;
    }
    a = a_next;
  }

  // Epilogue 1: bias + relu, park h rows in LDS as f16.
  // acc[t][j] = h[z = zt + j + hi*8][m = mbase + t*16 + col]
#pragma unroll
  for (int t = 0; t < 8; ++t) {
    int m = mbase + t * 16 + col;
    float bv = bias1[m];
#pragma unroll
    for (int j = 0; j < 8; ++j) {
      float h = acc[t][j] + bv;
      h = h > 0.f ? h : 0.f;
      lds_h[(j + hi * 8) * 1024 + m] = (_Float16)h;
    }
  }
  __syncthreads();

  // Epilogue 2: fc2 (1024 -> 10) for this block's 16 batch rows.
  if (tid < 160) {
    int zl = tid / 10, j = tid % 10;
    const float*    wrow = fc2w + j * 1024;
    const _Float16* hrow = lds_h + zl * 1024;
    float s = fc2b[j];
#pragma unroll 4
    for (int m = 0; m < 1024; m += 4) {
      s += (float)hrow[m + 0] * wrow[m + 0]
         + (float)hrow[m + 1] * wrow[m + 1]
         + (float)hrow[m + 2] * wrow[m + 2]
         + (float)hrow[m + 3] * wrow[m + 3];
    }
    out[(zt + zl) * 10 + j] = s;
  }
}

extern "C" void kernel_launch(void* const* d_in, const int* in_sizes, int n_in,
                              void* d_out, int out_size, void* d_ws, size_t ws_size,
                              hipStream_t stream) {
  const float* x     = (const float*)d_in[0];
  const float* u1    = (const float*)d_in[1];
  const float* u2    = (const float*)d_in[2];
  const float* u3    = (const float*)d_in[3];
  const float* b2    = (const float*)d_in[4];
  const float* b1    = (const float*)d_in[5];
  const float* bias1 = (const float*)d_in[6];
  const float* fc2w  = (const float*)d_in[7];
  const float* fc2b  = (const float*)d_in[8];
  float* out = (float*)d_out;

  char* ws = (char*)d_ws;
  float*    C  = (float*)(ws);                                   //  16 KB
  float*    T1 = (float*)(ws + (16u << 10));                     //  64 KB
  float*    T2 = (float*)(ws + (80u << 10));                     // 256 KB
  _Float16* Wh = (_Float16*)(ws + (336u << 10));                 //   2 MB
  _Float16* Xh = (_Float16*)(ws + (336u << 10) + (2u << 20));    //   2 MB

  k_core<<<16,   256, 0, stream>>>(b2, b1, C);
  k_t1  <<<64,   256, 0, stream>>>(u1, C, T1);
  k_t2  <<<256,  256, 0, stream>>>(T1, u2, T2);
  k_w   <<<4096, 256, 0, stream>>>(T2, u3, Wh);
  k_cvtx<<<4096, 256, 0, stream>>>(x, Xh);
  k_gemm<<<64,   256, 0, stream>>>(Xh, Wh, bias1, fc2w, fc2b, out);
}
